// SentimentAnalyzer_30485677867719
// MI455X (gfx1250) — matrix operand order repaired and checked by reading, not compiled, not run on hardware
//
#include <hip/hip_runtime.h>
#include <stdint.h>

// ---------------------------------------------------------------------------
// LSTM (S=2048, B=64, E=512, H=1024) + log_softmax for MI455X (gfx1250).
//   * bf16 WMMA 16x16x32, f32 accumulation
//   * persistent kernel, 32 WGs (each owns 32 hidden cols), grid barrier/step
//   * h(t) staged in LDS per step; x(t+1) PREFETCHED via async global->LDS
//     (double-buffered) so the copy drains behind the whole step's compute;
//     input-projection GEMM runs first to hide the h-copy latency, then
//     s_wait_asynccnt(16) waits only on the h chunks (ASYNCcnt is in-order)
//   * B-frags read once per WG from pre-packed bf16 fragment tiles with
//     non-temporal hint (384KB/step streams through L2, no L0 thrash)
//   * LDS: 300KB of the 320KB WGP budget (h 129K + x 2x65K + gates 33K + c 8K)
// ---------------------------------------------------------------------------

typedef __attribute__((ext_vector_type(16))) __bf16 v16bf;
typedef __attribute__((ext_vector_type(8)))  float  v8f;
typedef int v4i_t __attribute__((vector_size(16)));

#define S_LEN 2048
#define BATCH 64
#define EDIM  512
#define HDIM  1024
#define GDIM  4096   // 4*HDIM
#define NWG   32
#define COLS  32     // hidden columns owned per WG
#define TPB   256

// LDS layout (halfword / float units)
#define HPITCH 1032              // 1024 + 8 pad  (516 dwords % 64 = 4)
#define XPITCH 520               // 512 + 8 pad   (260 dwords % 64 = 4)
#define GPITCH 132               // 128 gate cols + 4 pad (f32)
#define XBYTES (BATCH * XPITCH * 2)                          // 66560
#define SM_H_OFF 0
#define SM_X_OFF (BATCH * HPITCH * 2)                        // 132096
#define SM_G_OFF (SM_X_OFF + 2 * XBYTES)                     // 265216
#define SM_C_OFF (SM_G_OFF + BATCH * GPITCH * 4)             // 299008
#define SM_BYTES (SM_C_OFF + BATCH * COLS * 4)               // 307200

union FragBF { uint4 q[2]; v4i_t w[2]; v16bf v; };

// ---- async global->LDS (CDNA5 GLOBAL_LOAD_ASYNC_TO_LDS_B128) --------------
#ifndef HAVE_ASYNC_LDS
#if defined(__has_builtin)
#if __has_builtin(__builtin_amdgcn_global_load_async_to_lds_b128)
#define HAVE_ASYNC_LDS 1
#endif
#endif
#endif
#ifndef HAVE_ASYNC_LDS
#define HAVE_ASYNC_LDS 0
#endif

__device__ __forceinline__ void cp16_to_lds(void* lds_dst, const void* gsrc) {
#if HAVE_ASYNC_LDS
  __builtin_amdgcn_global_load_async_to_lds_b128(
      (v4i_t __attribute__((address_space(1)))*)gsrc,
      (v4i_t __attribute__((address_space(3)))*)lds_dst, 0, 0);
#else
  *(uint4*)lds_dst = *(const uint4*)gsrc;
#endif
}
// wait until <= 16 outstanding: covers the 32 h-copy instructions issued
// before the 16 x-prefetch instructions (per-wave ASYNCcnt is in-order)
__device__ __forceinline__ void cp_lds_wait_h() {
#if HAVE_ASYNC_LDS
#if defined(__has_builtin) && __has_builtin(__builtin_amdgcn_s_wait_asynccnt)
  __builtin_amdgcn_s_wait_asynccnt(16);
#else
  asm volatile("s_wait_asynccnt 0x10" ::: "memory");
#endif
#endif
}
__device__ __forceinline__ void cp_lds_wait0() {
#if HAVE_ASYNC_LDS
#if defined(__has_builtin) && __has_builtin(__builtin_amdgcn_s_wait_asynccnt)
  __builtin_amdgcn_s_wait_asynccnt(0);
#else
  asm volatile("s_wait_asynccnt 0x0" ::: "memory");
#endif
#endif
}

__device__ __forceinline__ unsigned short f2bf(float f) {
  unsigned u = __float_as_uint(f);
  u += 0x7FFFu + ((u >> 16) & 1u);       // round-to-nearest-even
  return (unsigned short)(u >> 16);
}

// ---------------------------------------------------------------------------
// prep: zero h double-buffer + barrier counter, build combined bias
// ---------------------------------------------------------------------------
__global__ void prep_kernel(const float* __restrict__ b_ih,
                            const float* __restrict__ b_hh,
                            float* __restrict__ bias,
                            unsigned short* __restrict__ hbuf,
                            unsigned* __restrict__ bar) {
  int gid = blockIdx.x * TPB + threadIdx.x;
  if (gid < GDIM) bias[gid] = b_ih[gid] + b_hh[gid];
  if (gid < 2 * BATCH * HDIM) hbuf[gid] = 0;   // h0 = 0 (bf16 zero)
  if (gid == 0) *bar = 0;                      // reset grid barrier
}

// ---------------------------------------------------------------------------
// pack_w: fp32 W[4096,K] row-major -> bf16 WMMA-B fragment tiles.
//   elem j of lane L = W[ntile*16 + (L&15)][kc*32 + (L>>4)*8 + j + (j>=8)*8]
// ---------------------------------------------------------------------------
__global__ void pack_w_kernel(const float* __restrict__ W,
                              unsigned short* __restrict__ out, int K) {
  int gid  = blockIdx.x * TPB + threadIdx.x;
  int lane = gid & 31;
  int tk   = gid >> 5;              // ntile*(K/32) + kc
  int KC   = K >> 5;
  int ntile = tk / KC;
  int kc    = tk - ntile * KC;
  const float* src = W + (size_t)(ntile * 16 + (lane & 15)) * K
                       + kc * 32 + (lane >> 4) * 8;
  unsigned short* dst = out + (size_t)tk * 512 + lane * 16;
#pragma unroll
  for (int j = 0; j < 16; ++j) dst[j] = f2bf(src[j + (j >= 8 ? 8 : 0)]);
}

// ---------------------------------------------------------------------------
// embed: gather + fp32->bf16 for all timesteps: xq[s][b][e]
// ---------------------------------------------------------------------------
__global__ void embed_kernel(const int* __restrict__ X,
                             const float* __restrict__ emb,
                             unsigned short* __restrict__ xq) {
  size_t gid = (size_t)blockIdx.x * TPB + threadIdx.x;  // S*B*E threads
  int    e   = (int)(gid & (EDIM - 1));
  size_t sb  = gid >> 9;                                // E == 512
  int    tok = X[sb];
  xq[gid] = f2bf(emb[(size_t)tok * EDIM + e]);
}

// ---------------------------------------------------------------------------
// persistent LSTM kernel: 32 WGs x 256 threads, grid barrier per step
// wave w (0..7): n-tile = w; gate q = w>>1; col16 = w&1.
// Each wave holds 4 accumulators (all M tiles) so B-frags load exactly once.
// ---------------------------------------------------------------------------
__global__ __launch_bounds__(TPB) void lstm_persistent(
    const unsigned short* __restrict__ xq,     // [S][B][E] bf16
    const unsigned short* __restrict__ whh_t,  // packed B-frags, K=1024
    const unsigned short* __restrict__ wih_t,  // packed B-frags, K=512
    const float* __restrict__ bias,            // [4096] f32 (b_ih+b_hh)
    unsigned short* __restrict__ hbuf,         // 2 x [B][H] bf16
    float* __restrict__ hfinal,                // [B][H] f32
    unsigned* __restrict__ bar) {
  extern __shared__ char smem[];
  unsigned short* hsh = (unsigned short*)(smem + SM_H_OFF);
  unsigned short* xsh = (unsigned short*)(smem + SM_X_OFF);  // 2 buffers
  float*          gsh = (float*)(smem + SM_G_OFF);
  float*          csh = (float*)(smem + SM_C_OFF);

  const int tid  = threadIdx.x;
  const int lane = tid & 31;
  const int wave = tid >> 5;
  const int cg   = blockIdx.x;                 // owns h cols [cg*32, cg*32+32)

  const int q     = wave >> 1;                 // gate 0..3
  const int col16 = wave & 1;                  // which 16-col half
  const int ntile_hh = q * 64 + cg * 2 + col16;

  // A-fragment lane addressing (16-bit 16x32 A layout)
  const int alr = lane & 15;                   // M row within tile
  const int ah8 = (lane >> 4) * 8;             // K sub-offset

  const unsigned short* bhh = whh_t + (size_t)ntile_hh * 32 * 512 + lane * 16;
  const unsigned short* bih = wih_t + (size_t)ntile_hh * 16 * 512 + lane * 16;

  for (int i = tid; i < BATCH * COLS; i += TPB) csh[i] = 0.0f;

  // ---- prologue: stage x(0) into x buffer 0 ----
  for (int i = tid; i < BATCH * (EDIM / 8); i += TPB) {        // 64 x 64
    int r = i >> 6, ch = (i & 63) * 8;
    cp16_to_lds(&xsh[r * XPITCH + ch], &xq[r * EDIM + ch]);
  }
  cp_lds_wait0();
  __syncthreads();

  for (int t = 0; t < S_LEN; ++t) {
    const unsigned short* hcur = hbuf + (size_t)(t & 1) * (BATCH * HDIM);
    unsigned short*       hnxt = hbuf + (size_t)((t + 1) & 1) * (BATCH * HDIM);
    const unsigned short* xnow = xsh + (size_t)(t & 1) * (BATCH * XPITCH);
    unsigned short*      xnext = xsh + (size_t)((t + 1) & 1) * (BATCH * XPITCH);

    // ---- issue async h(t) -> LDS (32 instructions/wave) ----
    for (int i = tid; i < BATCH * (HDIM / 8); i += TPB) {      // 64 x 128
      int r = i >> 7, ch = (i & 127) * 8;
      cp16_to_lds(&hsh[r * HPITCH + ch], &hcur[r * HDIM + ch]);
    }
    // ---- issue async prefetch x(t+1) -> other LDS buffer (16/wave) ----
    if (t + 1 < S_LEN) {
      const unsigned short* xsrc = xq + (size_t)(t + 1) * (BATCH * EDIM);
      for (int i = tid; i < BATCH * (EDIM / 8); i += TPB) {    // 64 x 64
        int r = i >> 6, ch = (i & 63) * 8;
        cp16_to_lds(&xnext[r * XPITCH + ch], &xsrc[r * EDIM + ch]);
      }
    }

    v8f acc[4];
#pragma unroll
    for (int m = 0; m < 4; ++m) acc[m] = (v8f){0.f,0.f,0.f,0.f,0.f,0.f,0.f,0.f};

    FragBF a, b;
    // ---- input projection FIRST (x staged last step; hides h copy) ----
    {
      const unsigned short* bp = bih;
#pragma unroll 2
      for (int kc = 0; kc < EDIM / 32; ++kc) {
        b.w[0] = __builtin_nontemporal_load((const v4i_t*)(bp));
        b.w[1] = __builtin_nontemporal_load((const v4i_t*)(bp + 8));
        const int abase = kc * 32 + ah8;
#pragma unroll
        for (int m = 0; m < 4; ++m) {
          const unsigned short* ap = &xnow[(m * 16 + alr) * XPITCH + abase];
          a.q[0] = *(const uint4*)(ap);
          a.q[1] = *(const uint4*)(ap + 16);
          acc[m] = __builtin_amdgcn_wmma_f32_16x16x32_bf16(
              false, a.v, false, b.v, (short)0, acc[m], false, false);
        }
        bp += 512;
      }
    }

    // h(t) copy done? (in-order ASYNCcnt: 32 h ops precede 16 x ops)
    cp_lds_wait_h();
    __syncthreads();

    // ---- recurrent projection: h(t) x Whh^T tile-column ----
    {
      const unsigned short* bp = bhh;
#pragma unroll 2
      for (int kc = 0; kc < HDIM / 32; ++kc) {
        b.w[0] = __builtin_nontemporal_load((const v4i_t*)(bp));
        b.w[1] = __builtin_nontemporal_load((const v4i_t*)(bp + 8));
        const int abase = kc * 32 + ah8;
#pragma unroll
        for (int m = 0; m < 4; ++m) {
          const unsigned short* ap = &hsh[(m * 16 + alr) * HPITCH + abase];
          a.q[0] = *(const uint4*)(ap);
          a.q[1] = *(const uint4*)(ap + 16);
          acc[m] = __builtin_amdgcn_wmma_f32_16x16x32_bf16(
              false, a.v, false, b.v, (short)0, acc[m], false, false);
        }
        bp += 512;
      }
    }

    // ---- scatter accumulators to gate LDS (C/D layout: VGPR v -> row) ----
    {
      const int ccol  = q * 32 + col16 * 16 + (lane & 15);
      const int rbase = (lane >> 4) * 8;
#pragma unroll
      for (int m = 0; m < 4; ++m)
#pragma unroll
        for (int v = 0; v < 8; ++v)
          gsh[(m * 16 + rbase + v) * GPITCH + ccol] = acc[m][v];
    }
    __syncthreads();

    // ---- elementwise LSTM cell update for this WG's 32 columns ----
    for (int it = tid; it < BATCH * COLS; it += TPB) {
      const int r    = it >> 5;
      const int col  = it & 31;
      const int gcol = cg * COLS + col;
      float gi = gsh[r * GPITCH +          col] + bias[           gcol];
      float gf = gsh[r * GPITCH + COLS   + col] + bias[    HDIM + gcol];
      float gg = gsh[r * GPITCH + 2*COLS + col] + bias[2 * HDIM + gcol];
      float go = gsh[r * GPITCH + 3*COLS + col] + bias[3 * HDIM + gcol];
      float ig = 1.0f / (1.0f + __expf(-gi));
      float fg = 1.0f / (1.0f + __expf(-gf));
      float gv = tanhf(gg);
      float og = 1.0f / (1.0f + __expf(-go));
      float c  = fg * csh[it] + ig * gv;
      csh[it]  = c;
      float h  = og * tanhf(c);
      hnxt[r * HDIM + gcol] = f2bf(h);
      if (t == S_LEN - 1) hfinal[r * HDIM + gcol] = h;
    }
    __syncthreads();

    // drain x(t+1) prefetch (had the whole step to complete)
    cp_lds_wait0();

    // ---- grid-wide barrier (monotonic arrive counter) ----
    if (tid == 0) {
      __threadfence();
      __hip_atomic_fetch_add(bar, 1u, __ATOMIC_RELEASE, __HIP_MEMORY_SCOPE_AGENT);
      const unsigned target = (unsigned)(t + 1) * NWG;
      while (__hip_atomic_load(bar, __ATOMIC_ACQUIRE, __HIP_MEMORY_SCOPE_AGENT) < target)
        __builtin_amdgcn_s_sleep(2);
    }
    __syncthreads();
  }
}

// ---------------------------------------------------------------------------
// log_softmax over H=1024 per batch row
// ---------------------------------------------------------------------------
__global__ void logsoftmax_kernel(const float* __restrict__ h,
                                  float* __restrict__ out) {
  __shared__ float red[TPB];
  const int r   = blockIdx.x;
  const int tid = threadIdx.x;
  const float* row = h + (size_t)r * HDIM;

  float mx = -3.4e38f;
  for (int j = tid; j < HDIM; j += TPB) mx = fmaxf(mx, row[j]);
  red[tid] = mx; __syncthreads();
  for (int s = TPB / 2; s > 0; s >>= 1) {
    if (tid < s) red[tid] = fmaxf(red[tid], red[tid + s]);
    __syncthreads();
  }
  mx = red[0]; __syncthreads();

  float sum = 0.0f;
  for (int j = tid; j < HDIM; j += TPB) sum += __expf(row[j] - mx);
  red[tid] = sum; __syncthreads();
  for (int s = TPB / 2; s > 0; s >>= 1) {
    if (tid < s) red[tid] += red[tid + s];
    __syncthreads();
  }
  const float lse = logf(red[0]);
  __syncthreads();

  for (int j = tid; j < HDIM; j += TPB)
    out[(size_t)r * HDIM + j] = row[j] - mx - lse;
}

// ---------------------------------------------------------------------------
extern "C" void kernel_launch(void* const* d_in, const int* in_sizes, int n_in,
                              void* d_out, int out_size, void* d_ws, size_t ws_size,
                              hipStream_t stream) {
  (void)in_sizes; (void)n_in; (void)out_size; (void)ws_size;
  const int*   X    = (const int*)d_in[0];    // [S,B] token ids
  const float* emb  = (const float*)d_in[1];  // [VOCAB,E]
  const float* W_ih = (const float*)d_in[2];  // [4H,E]
  const float* W_hh = (const float*)d_in[3];  // [4H,H]
  const float* b_ih = (const float*)d_in[4];  // [4H]
  const float* b_hh = (const float*)d_in[5];  // [4H]
  float*       out  = (float*)d_out;          // [B,H]

  // workspace layout (256B aligned)
  char* ws = (char*)d_ws;
  size_t off = 0;
  auto alloc = [&](size_t bytes) {
    size_t o = off;
    off = (off + bytes + 255) & ~(size_t)255;
    return o;
  };
  unsigned short* whh_t  = (unsigned short*)(ws + alloc((size_t)GDIM * HDIM * 2));
  unsigned short* wih_t  = (unsigned short*)(ws + alloc((size_t)GDIM * EDIM * 2));
  float*          bias   = (float*)(ws + alloc((size_t)GDIM * 4));
  unsigned short* hbuf   = (unsigned short*)(ws + alloc((size_t)2 * BATCH * HDIM * 2));
  float*          hfinal = (float*)(ws + alloc((size_t)BATCH * HDIM * 4));
  unsigned*       bar    = (unsigned*)(ws + alloc(256));
  unsigned short* xq     = (unsigned short*)(ws + alloc((size_t)S_LEN * BATCH * EDIM * 2));

  // 1) init barrier/h0 + combined bias
  prep_kernel<<<(2 * BATCH * HDIM + TPB - 1) / TPB, TPB, 0, stream>>>(
      b_ih, b_hh, bias, hbuf, bar);

  // 2) pack weights to WMMA B-fragment layout (bf16)
  {
    int threads_hh = (GDIM / 16) * (HDIM / 32) * 32;  // 262144
    pack_w_kernel<<<threads_hh / TPB, TPB, 0, stream>>>(W_hh, whh_t, HDIM);
    int threads_ih = (GDIM / 16) * (EDIM / 32) * 32;  // 131072
    pack_w_kernel<<<threads_ih / TPB, TPB, 0, stream>>>(W_ih, wih_t, EDIM);
  }

  // 3) embedding gather + bf16 convert for all timesteps
  {
    size_t total = (size_t)S_LEN * BATCH * EDIM;      // 67,108,864
    embed_kernel<<<(unsigned)(total / TPB), TPB, 0, stream>>>(X, emb, xq);
  }

  // 4) persistent recurrent loop (grid barrier per step, 300KB dynamic LDS)
  lstm_persistent<<<NWG, TPB, SM_BYTES, stream>>>(xq, whh_t, wih_t, bias,
                                                  hbuf, hfinal, bar);

  // 5) log_softmax on final hidden state
  logsoftmax_kernel<<<BATCH, TPB, 0, stream>>>(hfinal, out);
}
